// AttentionBlock_62517543960698
// MI455X (gfx1250) — matrix operand order repaired and checked by reading, not compiled, not run on hardware
//
#include <hip/hip_runtime.h>

typedef __bf16 bf16;
typedef __attribute__((ext_vector_type(16))) __bf16 v16bf;
typedef __attribute__((ext_vector_type(8)))  float  v8f;

#define BATCH 8
#define CH    256
#define NPIX  4096
#define NGRP  8
#define CPG   32
#define GN_EPS    1e-5f
#define ATT_SCALE 0.0625f   // 1/sqrt(256)
#define BJ 64               // key/value tile (j) per iteration
#define NITER (NPIX / BJ)

union V16U { v16bf v; uint4 u4[2]; };

__device__ __forceinline__ v8f wmma_bf16(v16bf a, v16bf b, v8f c) {
    // D = A(16x32 bf16) * B(32x16 bf16) + C(16x16 f32)
    return __builtin_amdgcn_wmma_f32_16x16x32_bf16(false, a, false, b, (short)0, c, false, false);
}

// ---- CDNA5 async global->LDS copy (ASYNCcnt path), GV addressing mode ----
__device__ __forceinline__ void async_load16(const bf16* gsrc, bf16* ldst) {
    unsigned loff = (unsigned)(uintptr_t)ldst;               // LDS aperture: addr[31:0] = LDS byte address
    unsigned long long ga = (unsigned long long)(uintptr_t)gsrc;
    asm volatile("global_load_async_to_lds_b128 %0, %1, off"
                 :: "v"(loff), "v"(ga) : "memory");
}
#if __has_builtin(__builtin_amdgcn_s_wait_asynccnt)
#define WAIT_ASYNC(n) __builtin_amdgcn_s_wait_asynccnt(n)
#else
#define WAIT_ASYNC_STR(n) asm volatile("s_wait_asynccnt " #n ::: "memory")
#define WAIT_ASYNC(n) WAIT_ASYNC_STR(n)
#endif

__device__ __forceinline__ float redmax16(float v) {
#pragma unroll
    for (int m = 8; m >= 1; m >>= 1) v = fmaxf(v, __shfl_xor(v, m, 32));
    return v;
}
__device__ __forceinline__ float redsum16(float v) {
#pragma unroll
    for (int m = 8; m >= 1; m >>= 1) v += __shfl_xor(v, m, 32);
    return v;
}

// ---------------- weight fp32 -> bf16 ----------------
__global__ void convert_bf16_kernel(const float* __restrict__ src, bf16* __restrict__ dst, int n) {
    int i = blockIdx.x * blockDim.x + threadIdx.x;
    if (i < n) dst[i] = (bf16)src[i];
}

// ---------------- groupnorm stats: one WG per (b,g) ----------------
__global__ __launch_bounds__(256) void gn_stats_kernel(const float* __restrict__ x,
                                                       float* __restrict__ stat) {
    __shared__ float s_sum[256];
    __shared__ float s_sq[256];
    int bg = blockIdx.x;               // 0..63
    int b = bg / NGRP, g = bg % NGRP;
    const float* p = x + ((size_t)b * CH + (size_t)g * CPG) * NPIX;
    float sum = 0.f, sq = 0.f;
    for (int i = threadIdx.x; i < CPG * NPIX; i += 256) {
        float v = p[i]; sum += v; sq += v * v;
    }
    s_sum[threadIdx.x] = sum; s_sq[threadIdx.x] = sq;
    __syncthreads();
    for (int s = 128; s > 0; s >>= 1) {
        if (threadIdx.x < s) {
            s_sum[threadIdx.x] += s_sum[threadIdx.x + s];
            s_sq[threadIdx.x]  += s_sq[threadIdx.x + s];
        }
        __syncthreads();
    }
    if (threadIdx.x == 0) {
        const float inv_n = 1.0f / (float)(CPG * NPIX);
        float mean = s_sum[0] * inv_n;
        float var  = s_sq[0] * inv_n - mean * mean;
        stat[bg * 2 + 0] = mean;
        stat[bg * 2 + 1] = rsqrtf(var + GN_EPS);
    }
}

// ------- groupnorm apply + transpose: x[b][c][n] f32 -> nx_t[b][n][c] bf16 -------
__global__ __launch_bounds__(256) void gn_apply_t_kernel(const float* __restrict__ x,
                                                         const float* __restrict__ stat,
                                                         const float* __restrict__ gn_w,
                                                         const float* __restrict__ gn_b,
                                                         bf16* __restrict__ nx_t) {
    __shared__ float tile[64][65];
    int b  = blockIdx.z;
    int n0 = blockIdx.x * 64;
    int c0 = blockIdx.y * 64;
    int t  = threadIdx.x;
    int ln = t % 64, lr = t / 64;
#pragma unroll
    for (int k = 0; k < 16; ++k) {
        int cl = lr + 4 * k;
        int c  = c0 + cl;
        int g  = c / CPG;
        float mean = stat[(b * NGRP + g) * 2 + 0];
        float rstd = stat[(b * NGRP + g) * 2 + 1];
        float v = x[((size_t)b * CH + c) * NPIX + n0 + ln];
        tile[cl][ln] = (v - mean) * rstd * gn_w[c] + gn_b[c];
    }
    __syncthreads();
#pragma unroll
    for (int k = 0; k < 16; ++k) {
        int nl = lr + 4 * k;
        nx_t[((size_t)b * NPIX + n0 + nl) * CH + c0 + ln] = (bf16)tile[ln][nl];
    }
}

// ---------------- QKV GEMM (bf16 WMMA, f32 accumulate) ----------------
// q_ws[b][n][c] (scaled by ATT_SCALE), k_ws[b][n][c], v_ws[b][c][n]
__global__ __launch_bounds__(256) void qkv_gemm_kernel(const bf16* __restrict__ nx_t,
                                                       const bf16* __restrict__ w_bf,
                                                       const float* __restrict__ bias,
                                                       bf16* __restrict__ q_ws,
                                                       bf16* __restrict__ k_ws,
                                                       bf16* __restrict__ v_ws) {
    int b    = blockIdx.z;
    int o0   = blockIdx.y * 16;               // 0..752 over 768 outputs
    int wave = threadIdx.x >> 5;
    int lane = threadIdx.x & 31;
    int ln   = lane & 15;
    int hf   = lane >> 4;
    int n0   = (blockIdx.x * 8 + wave) * 16;
    const bf16* nx = nx_t + (size_t)b * NPIX * CH;
    v8f acc = {};
    if (o0 < 512) {
        // q/k: transposed tile (rows=n, cols=o): A=nx_t rows, B=w rows as K-runs
        const bf16* arow = nx  + (size_t)(n0 + ln) * CH;
        const bf16* brow = w_bf + (size_t)(o0 + ln) * CH;
#pragma unroll
        for (int cb = 0; cb < CH; cb += 32) {
            V16U a;
            a.u4[0] = *(const uint4*)(arow + cb + hf * 16);
            a.u4[1] = *(const uint4*)(arow + cb + hf * 16 + 8);
            v16bf bm = *(const v16bf*)(brow + cb + hf * 16);
            acc = wmma_bf16(a.v, bm, acc);
        }
        float bia = bias[o0 + ln];
        bool isq  = (o0 < 256);
        bf16* dst = (isq ? q_ws : k_ws) + (size_t)b * NPIX * CH;
        int   oc  = o0 - (isq ? 0 : 256);
        float sc  = isq ? ATT_SCALE : 1.0f;
#pragma unroll
        for (int e = 0; e < 8; ++e) {
            int nr = n0 + e + hf * 8;
            dst[(size_t)nr * CH + oc + ln] = (bf16)((acc[e] + bia) * sc);
        }
    } else {
        // v: natural tile (rows=o, cols=n): A=w rows, B=nx_t rows as K-runs
        const bf16* arow = w_bf + (size_t)(o0 + ln) * CH;
        const bf16* brow = nx   + (size_t)(n0 + ln) * CH;
#pragma unroll
        for (int cb = 0; cb < CH; cb += 32) {
            V16U a;
            a.u4[0] = *(const uint4*)(arow + cb + hf * 16);
            a.u4[1] = *(const uint4*)(arow + cb + hf * 16 + 8);
            v16bf bm = *(const v16bf*)(brow + cb + hf * 16);
            acc = wmma_bf16(a.v, bm, acc);
        }
        int oc = o0 - 512;
        bf16* dst = v_ws + (size_t)b * CH * NPIX;
#pragma unroll
        for (int e = 0; e < 8; ++e) {
            int orow = oc + e + hf * 8;
            dst[(size_t)orow * NPIX + n0 + ln] = (bf16)(acc[e] + bias[512 + orow]);
        }
    }
}

// ---------------- flash attention: o_ws[b][i][c] = softmax(qk)v ----------------
// Double-buffered K/V tiles fed by GLOBAL_LOAD_ASYNC_TO_LDS (ASYNCcnt), 144 KB LDS.
__global__ __launch_bounds__(256) void attn_kernel(const bf16* __restrict__ q_ws,
                                                   const bf16* __restrict__ k_ws,
                                                   const bf16* __restrict__ v_ws,
                                                   bf16* __restrict__ o_ws) {
    __shared__ bf16 k_lds[2][BJ][CH];    // 2 x 32 KB, [j][c]
    __shared__ bf16 v_lds[2][CH][BJ];    // 2 x 32 KB, [c][j]
    __shared__ bf16 p_lds[8][16][BJ];    // 16 KB, per-wave P strip [i][j]
    int b    = blockIdx.y;
    int i0   = blockIdx.x * 128;         // 8 waves x 16 query rows
    int tid  = threadIdx.x;
    int wave = tid >> 5;
    int lane = tid & 31;
    int ln   = lane & 15;
    int hf   = lane >> 4;
    const bf16* qb = q_ws + (size_t)b * NPIX * CH;
    const bf16* kb = k_ws + (size_t)b * NPIX * CH;
    const bf16* vb = v_ws + (size_t)b * CH * NPIX;

    // resident Q fragments: 8 K-chunks of 32 channels
    V16U qf[8];
    {
        const bf16* qrow = qb + (size_t)(i0 + wave * 16 + ln) * CH;
#pragma unroll
        for (int c8 = 0; c8 < 8; ++c8) {
            qf[c8].u4[0] = *(const uint4*)(qrow + c8 * 32 + hf * 16);
            qf[c8].u4[1] = *(const uint4*)(qrow + c8 * 32 + hf * 16 + 8);
        }
    }
    v8f oacc[16] = {};
    float m[8], l[8];
#pragma unroll
    for (int e = 0; e < 8; ++e) { m[e] = -1e30f; l[e] = 0.f; }

    // async tile prefetch: K tile BJ x 256, V tile 256 x BJ; 16 b128 ops/thread
    auto prefetch = [&](int j0, int buf) {
        int krow = tid >> 2, kcol = (tid & 3) * 64;
        const bf16* ksrc = kb + (size_t)(j0 + krow) * CH + kcol;
        bf16* kdst = &k_lds[buf][krow][kcol];
#pragma unroll
        for (int i = 0; i < 8; ++i) async_load16(ksrc + 8 * i, kdst + 8 * i);
        const bf16* vsrc = vb + (size_t)tid * NPIX + j0;
        bf16* vdst = &v_lds[buf][tid][0];
#pragma unroll
        for (int i = 0; i < 8; ++i) async_load16(vsrc + 8 * i, vdst + 8 * i);
    };

    prefetch(0, 0);
    for (int it = 0; it < NITER; ++it) {
        int buf = it & 1;
        if (it + 1 < NITER) {           // overlap next tile's DMA with this tile's math
            prefetch((it + 1) * BJ, buf ^ 1);
            WAIT_ASYNC(16);             // the 16 just-issued may fly; current tile done
        } else {
            WAIT_ASYNC(0);
        }
        __syncthreads();                // all waves' tile data visible

        // S = Q Kt : four 16x16 j-subtiles, K-dim = 256 channels
        v8f s[4];
#pragma unroll
        for (int jt = 0; jt < 4; ++jt) {
            v8f sa = {};
            const bf16* brow = &k_lds[buf][jt * 16 + ln][0];
#pragma unroll
            for (int c8 = 0; c8 < 8; ++c8) {
                v16bf bm = *(const v16bf*)(brow + c8 * 32 + hf * 16);
                sa = wmma_bf16(qf[c8].v, bm, sa);
            }
            s[jt] = sa;
        }
        // online softmax (row = accumulator element e, cols across 16 lanes)
        float alpha[8];
#pragma unroll
        for (int e = 0; e < 8; ++e) {
            float mx = fmaxf(fmaxf(s[0][e], s[1][e]), fmaxf(s[2][e], s[3][e]));
            mx = redmax16(mx);
            float mn = fmaxf(m[e], mx);
            alpha[e] = __expf(m[e] - mn);
            m[e] = mn;
            float p0 = __expf(s[0][e] - mn);
            float p1 = __expf(s[1][e] - mn);
            float p2 = __expf(s[2][e] - mn);
            float p3 = __expf(s[3][e] - mn);
            s[0][e] = p0; s[1][e] = p1; s[2][e] = p2; s[3][e] = p3;
            float rs = redsum16((p0 + p1) + (p2 + p3));
            l[e] = l[e] * alpha[e] + rs;
        }
#pragma unroll
        for (int ct = 0; ct < 16; ++ct)
#pragma unroll
            for (int e = 0; e < 8; ++e) oacc[ct][e] *= alpha[e];
        // C-layout -> A-layout transpose of P through per-wave LDS strip
#pragma unroll
        for (int jt = 0; jt < 4; ++jt)
#pragma unroll
            for (int e = 0; e < 8; ++e)
                p_lds[wave][e + hf * 8][jt * 16 + ln] = (bf16)s[jt][e];
        // same-wave DS ordering: no barrier needed before re-reading own strip
#pragma unroll
        for (int jb = 0; jb < 2; ++jb) {    // two K=32 j-chunks
            V16U pa;
            const bf16* prow = &p_lds[wave][ln][jb * 32];
            pa.u4[0] = *(const uint4*)(prow + hf * 16);
            pa.u4[1] = *(const uint4*)(prow + hf * 16 + 8);
#pragma unroll
            for (int ct = 0; ct < 16; ++ct) {   // O += P Vt over 16 c-subtiles
                v16bf bm = *(const v16bf*)(&v_lds[buf][ct * 16 + ln][jb * 32 + hf * 16]);
                oacc[ct] = wmma_bf16(pa.v, bm, oacc[ct]);
            }
        }
        __syncthreads();                // reads of buf done before it gets overwritten next iter
    }
    // epilogue: normalize and store o_ws[b][i][c]
    bf16* ob = o_ws + (size_t)b * NPIX * CH;
    float inv[8];
#pragma unroll
    for (int e = 0; e < 8; ++e) inv[e] = 1.0f / l[e];
#pragma unroll
    for (int ct = 0; ct < 16; ++ct)
#pragma unroll
        for (int e = 0; e < 8; ++e) {
            int row = i0 + wave * 16 + e + hf * 8;
            ob[(size_t)row * CH + ct * 16 + ln] = (bf16)(oacc[ct][e] * inv[e]);
        }
}

// ---------------- output projection + bias + residual ----------------
__global__ __launch_bounds__(256) void proj_kernel(const bf16* __restrict__ o_ws,
                                                   const bf16* __restrict__ ow_bf,
                                                   const float* __restrict__ ob,
                                                   const float* __restrict__ x,
                                                   float* __restrict__ out) {
    int b    = blockIdx.z;
    int o0   = blockIdx.y * 16;
    int wave = threadIdx.x >> 5;
    int lane = threadIdx.x & 31;
    int ln   = lane & 15;
    int hf   = lane >> 4;
    int n0   = (blockIdx.x * 8 + wave) * 16;
    const bf16* arow = ow_bf + (size_t)(o0 + ln) * CH;
    const bf16* brow = o_ws + ((size_t)b * NPIX + n0 + ln) * CH;
    v8f acc = {};
#pragma unroll
    for (int cb = 0; cb < CH; cb += 32) {
        V16U a;
        a.u4[0] = *(const uint4*)(arow + cb + hf * 16);
        a.u4[1] = *(const uint4*)(arow + cb + hf * 16 + 8);
        v16bf bm = *(const v16bf*)(brow + cb + hf * 16);
        acc = wmma_bf16(a.v, bm, acc);
    }
#pragma unroll
    for (int e = 0; e < 8; ++e) {
        int orow = o0 + e + hf * 8;
        size_t idx = ((size_t)b * CH + orow) * NPIX + n0 + ln;
        out[idx] = acc[e] + ob[orow] + x[idx];
    }
}

extern "C" void kernel_launch(void* const* d_in, const int* in_sizes, int n_in,
                              void* d_out, int out_size, void* d_ws, size_t ws_size,
                              hipStream_t stream) {
    const float* x     = (const float*)d_in[0];
    const float* gn_w  = (const float*)d_in[1];
    const float* gn_b  = (const float*)d_in[2];
    const float* qkv_w = (const float*)d_in[3];
    const float* qkv_b = (const float*)d_in[4];
    const float* out_w = (const float*)d_in[5];
    const float* out_b = (const float*)d_in[6];
    float* out = (float*)d_out;

    char* ws = (char*)d_ws;
    size_t off = 0;
    float* stat    = (float*)(ws + off); off += 1024;
    bf16* qkvw_bf  = (bf16*)(ws + off);  off += (size_t)768 * 256 * 2;
    bf16* outw_bf  = (bf16*)(ws + off);  off += (size_t)256 * 256 * 2;
    bf16* nx_t     = (bf16*)(ws + off);  off += (size_t)BATCH * NPIX * CH * 2;
    bf16* q_ws     = (bf16*)(ws + off);  off += (size_t)BATCH * NPIX * CH * 2;
    bf16* k_ws     = (bf16*)(ws + off);  off += (size_t)BATCH * NPIX * CH * 2;
    bf16* v_ws     = (bf16*)(ws + off);  off += (size_t)BATCH * NPIX * CH * 2;
    bf16* o_ws     = (bf16*)(ws + off);  off += (size_t)BATCH * NPIX * CH * 2;

    convert_bf16_kernel<<<(768 * 256 + 255) / 256, 256, 0, stream>>>(qkv_w, qkvw_bf, 768 * 256);
    convert_bf16_kernel<<<(256 * 256 + 255) / 256, 256, 0, stream>>>(out_w, outw_bf, 256 * 256);
    gn_stats_kernel<<<64, 256, 0, stream>>>(x, stat);
    gn_apply_t_kernel<<<dim3(NPIX / 64, CH / 64, BATCH), 256, 0, stream>>>(x, stat, gn_w, gn_b, nx_t);
    qkv_gemm_kernel<<<dim3(NPIX / 16 / 8, 768 / 16, BATCH), 256, 0, stream>>>(nx_t, qkvw_bf, qkv_b,
                                                                              q_ws, k_ws, v_ws);
    attn_kernel<<<dim3(NPIX / 128, BATCH), 256, 0, stream>>>(q_ws, k_ws, v_ws, o_ws);
    proj_kernel<<<dim3(NPIX / 16 / 8, CH / 16, BATCH), 256, 0, stream>>>(o_ws, outw_bf, out_b, x, out);
}
